// test_GGCN_4861902979401
// MI455X (gfx1250) — matrix operand order repaired and checked by reading, not compiled, not run on hardware
//
#include <hip/hip_runtime.h>

typedef __attribute__((ext_vector_type(16))) __bf16 v16bf;
typedef __attribute__((ext_vector_type(8)))  __bf16 v8bf;
typedef __attribute__((ext_vector_type(8)))  float  v8f;

#define NNODES 10000
#define NGRAPH 64
#define TM 64          // block tile M
#define TN 64          // block tile N
#define TK 32          // k-step (WMMA K)
#define LDSS 40        // padded LDS row stride in bf16 (80 B: 16B-aligned, shifts banks)

static inline int ceil_div(int a, int b) { return (a + b - 1) / b; }

// ---------------------------------------------------------------------------
// NT WMMA GEMM: C[M,N] (f32) = A[M,K] (bf16 row-major) * BT[N,K]^T
// Both operands k-major => every transfer is b128 (global and LDS).
// 256 threads = 8 waves as 4(M) x 2(N); each wave owns a 16x32 strip:
//   2 x v_wmma_f32_16x16x32_bf16 per k-step sharing one A fragment.
// Software pipeline: registers -> ping-pong LDS, one barrier per k-step,
// global loads for tile k+1 overlapped with WMMA on tile k.
// ---------------------------------------------------------------------------
__global__ __launch_bounds__(256) void k_gemm_nt(
    const __bf16* __restrict__ A, const __bf16* __restrict__ BT,
    float* __restrict__ C, int M, int K, int N) {
  __shared__ __align__(16) __bf16 As[2][TM][LDSS];   // [buf][m][k]
  __shared__ __align__(16) __bf16 Bs[2][TN][LDSS];   // [buf][n][k]

  const int t    = threadIdx.x;
  const int lane = t & 31;
  const int wave = t >> 5;
  const int wm   = wave >> 1;        // 0..3
  const int wn   = wave & 1;         // 0..1
  const int m0   = blockIdx.x * TM;
  const int n0   = blockIdx.y * TN;
  const int hl   = lane >> 4;        // lane half
  const int l16  = lane & 15;

  // staging: each thread moves one 8-elem (16 B) chunk per matrix per k-step
  const int srow = t >> 2;           // 0..63
  const int skc  = (t & 3) << 3;     // 0,8,16,24

  // clamp tail rows instead of branching: garbage rows only feed guarded stores
  int ar = m0 + srow; if (ar >= M) ar = M - 1;
  const __bf16* aptr = A  + (size_t)ar * K + skc;
  const __bf16* bptr = BT + (size_t)(n0 + srow) * K + skc;   // N % 64 == 0 always

  const int nk = K / TK;             // K is always a multiple of 32
  v8f acc0 = {}, acc1 = {};

  // prefill tile 0 into registers
  v8bf ra = *(const v8bf*)(aptr);
  v8bf rb = *(const v8bf*)(bptr);

  for (int ki = 0; ki < nk; ++ki) {
    const int cur = ki & 1;
    *(v8bf*)&As[cur][srow][skc] = ra;
    *(v8bf*)&Bs[cur][srow][skc] = rb;
    __syncthreads();                  // single barrier per k-step (ping-pong LDS)

    if (ki + 1 < nk) {                // uniform scalar branch, no exec churn
      ra = *(const v8bf*)(aptr + (size_t)(ki + 1) * TK);
      rb = *(const v8bf*)(bptr + (size_t)(ki + 1) * TK);
      // speculative prefetch two tiles ahead (OOB prefetch is silently dropped)
      __builtin_prefetch(aptr + (size_t)(ki + 2) * TK, 0, 3);
      __builtin_prefetch(bptr + (size_t)(ki + 2) * TK, 0, 3);
    }

    // A fragment (16x32): lane half 0 -> K 0-7 / 16-23, half 1 -> K 8-15 / 24-31
    v16bf afrag;
    const int am = wm * 16 + l16;
#pragma unroll
    for (int j = 0; j < 8; ++j) {
      int kb = ((j < 4) ? 0 : 16) + hl * 8 + (j & 3) * 2;
      afrag[2 * j]     = As[cur][am][kb];
      afrag[2 * j + 1] = As[cur][am][kb + 1];
    }
    // B fragments (32x16 each): n = lane&15, lane half 0 -> K 0-7 / 16-23, half 1 -> K 8-15 / 24-31
    const int bn0 = wn * 32 + l16;
    v16bf bfrag0, bfrag1;
#pragma unroll
    for (int j = 0; j < 8; ++j) {
      int kb = ((j < 4) ? 0 : 16) + hl * 8 + (j & 3) * 2;
      bfrag0[2 * j]     = Bs[cur][bn0][kb];
      bfrag0[2 * j + 1] = Bs[cur][bn0][kb + 1];
      bfrag1[2 * j]     = Bs[cur][bn0 + 16][kb];
      bfrag1[2 * j + 1] = Bs[cur][bn0 + 16][kb + 1];
    }

    acc0 = __builtin_amdgcn_wmma_f32_16x16x32_bf16(
        false, afrag, false, bfrag0, (short)0, acc0, false, false);
    acc1 = __builtin_amdgcn_wmma_f32_16x16x32_bf16(
        false, afrag, false, bfrag1, (short)0, acc1, false, false);
    // no trailing barrier: next iter writes the other LDS buffer
  }

  // C/D layout: VGPR v -> M = v + 8*(lane>>4), N = lane&15
  const int gn0 = n0 + wn * 32 + l16;
#pragma unroll
  for (int v = 0; v < 8; ++v) {
    int gm = m0 + wm * 16 + v + hl * 8;
    if (gm < M) {
      C[(size_t)gm * N + gn0]      = acc0[v];
      C[(size_t)gm * N + gn0 + 16] = acc1[v];
    }
  }
}

// ---------------------------------------------------------------------------
// Elementwise helpers
// ---------------------------------------------------------------------------
__global__ void k_copy_f32(const float* __restrict__ in, float* __restrict__ out, int n) {
  int i = blockIdx.x * blockDim.x + threadIdx.x;
  if (i < n) out[i] = in[i];
}

__global__ void k_cast_bf16(const float* __restrict__ in, __bf16* __restrict__ out, int n) {
  int i = blockIdx.x * blockDim.x + threadIdx.x;
  if (i < n) out[i] = (__bf16)in[i];
}

// in: [R][Ccols] f32 -> out: [Ccols][R] bf16 (transpose + cast; for w[l] slices)
__global__ void k_castT_bf16(const float* __restrict__ in, __bf16* __restrict__ out,
                             int R, int Ccols) {
  int i = blockIdx.x * blockDim.x + threadIdx.x;
  if (i >= R * Ccols) return;
  int r = i / Ccols, c = i % Ccols;
  out[(size_t)c * R + r] = (__bf16)in[i];
}

__global__ void k_fill(float* __restrict__ p, int n, float v) {
  int i = blockIdx.x * blockDim.x + threadIdx.x;
  if (i < n) p[i] = v;
}

// a[dst[e]] += m[src[e]], 4 features per thread; atomics resolve in L2
__global__ void k_scatter_add(const float* __restrict__ m, const int* __restrict__ src,
                              const int* __restrict__ dst, float* __restrict__ a,
                              int E, int C) {
  int nvec = C >> 2;
  int i = blockIdx.x * blockDim.x + threadIdx.x;
  if (i >= E * nvec) return;
  int e  = i / nvec;
  int f4 = (i - e * nvec) << 2;
  int s = src[e], d = dst[e];
  const float* mp = m + (size_t)s * C + f4;
  float*       ap = a + (size_t)d * C + f4;
  atomicAdd(ap + 0, mp[0]);
  atomicAdd(ap + 1, mp[1]);
  atomicAdd(ap + 2, mp[2]);
  atomicAdd(ap + 3, mp[3]);
}

// fused GRUCell update (biases added here; gi/gh are bias-free GEMM outputs)
__global__ void k_gru(const float* __restrict__ gi, const float* __restrict__ gh,
                      const float* __restrict__ bih, const float* __restrict__ bhh,
                      const float* __restrict__ h, float* __restrict__ hout,
                      int Nn, int C) {
  int i = blockIdx.x * blockDim.x + threadIdx.x;
  if (i >= Nn * C) return;
  int node = i / C, c = i - node * C;
  const float* gir = gi + (size_t)node * 3 * C;
  const float* ghr = gh + (size_t)node * 3 * C;
  float ir  = gir[c]         + bih[c];
  float iz  = gir[C + c]     + bih[C + c];
  float inn = gir[2 * C + c] + bih[2 * C + c];
  float hr  = ghr[c]         + bhh[c];
  float hz  = ghr[C + c]     + bhh[C + c];
  float hn  = ghr[2 * C + c] + bhh[2 * C + c];
  float r = 1.0f / (1.0f + __expf(-(ir + hr)));
  float z = 1.0f / (1.0f + __expf(-(iz + hz)));
  float n = tanhf(inn + r * hn);
  hout[i] = (1.0f - z) * n + z * h[i];
}

// relu(h1[N,128]) padded with zeros into h2[N,256]
__global__ void k_relu_pad(const float* __restrict__ h1, float* __restrict__ h2, int Nn) {
  int i = blockIdx.x * blockDim.x + threadIdx.x;
  if (i >= Nn * 256) return;
  int node = i >> 8, c = i & 255;
  h2[i] = (c < 128) ? fmaxf(h1[node * 128 + c], 0.0f) : 0.0f;
}

// segment-max via CAS float-max
__global__ void k_segmax(const float* __restrict__ h, const int* __restrict__ batch,
                         float* __restrict__ ymax, int Nn, int C) {
  int i = blockIdx.x * blockDim.x + threadIdx.x;
  if (i >= Nn * C) return;
  int node = i / C, c = i - node * C;
  float v = h[i];
  unsigned int* addr = (unsigned int*)&ymax[(size_t)batch[node] * C + c];
  unsigned int old = *addr;
  while (v > __uint_as_float(old)) {
    unsigned int assumed = old;
    old = atomicCAS(addr, assumed, __float_as_uint(v));
    if (old == assumed) break;
  }
}

// out[g][o] = ymax[g,:] . fc_w[o,:] + fc_b[o]   (64x6, trivial)
__global__ void k_fc(const float* __restrict__ y, const float* __restrict__ w,
                     const float* __restrict__ b, float* __restrict__ out) {
  int i = blockIdx.x * blockDim.x + threadIdx.x;
  if (i >= NGRAPH * 6) return;
  int g = i / 6, o = i - g * 6;
  float s = b[o];
  for (int k = 0; k < 256; ++k) s += y[g * 256 + k] * w[o * 256 + k];
  out[i] = s;
}

// ---------------------------------------------------------------------------
// One GatedGraphConv phase (3 layers).
// All GEMMs are NT: B operands are stored [n][k].
//   m  = h @ w[l]      -> BT = w[l]^T   (pre-transposed tiny C x C)
//   gi = a @ wih.T     -> BT = wih      (already [n][k]! plain cast)
//   gh = h @ whh.T     -> BT = whh      (already [n][k]! plain cast)
// ---------------------------------------------------------------------------
static void run_phase(int C, const float* w, const float* wih, const float* whh,
                      const float* bih, const float* bhh,
                      const int* src, const int* dst, int E,
                      float** ph, float** phn,
                      float* a, float* m, float* gi, float* gh,
                      __bf16* hb, __bf16* ab,
                      __bf16* wbW, __bf16* wbIH, __bf16* wbHH,
                      hipStream_t stream) {
  const int G3 = 3 * C;
  const int NB = 256;
  for (int l = 0; l < 3; ++l)
    k_castT_bf16<<<ceil_div(C * C, NB), NB, 0, stream>>>(
        w + (size_t)l * C * C, wbW + (size_t)l * C * C, C, C);
  k_cast_bf16<<<ceil_div(G3 * C, NB), NB, 0, stream>>>(wih, wbIH, G3 * C);
  k_cast_bf16<<<ceil_div(G3 * C, NB), NB, 0, stream>>>(whh, wbHH, G3 * C);

  dim3 gC(ceil_div(NNODES, TM), C / TN);
  dim3 gG(ceil_div(NNODES, TM), G3 / TN);
  float* h  = *ph;
  float* hn = *phn;
  const int nvec = C >> 2;

  for (int l = 0; l < 3; ++l) {
    k_cast_bf16<<<ceil_div(NNODES * C, NB), NB, 0, stream>>>(h, hb, NNODES * C);
    k_gemm_nt<<<gC, NB, 0, stream>>>(hb, wbW + (size_t)l * C * C, m, NNODES, C, C);
    k_fill<<<ceil_div(NNODES * C, NB), NB, 0, stream>>>(a, NNODES * C, 0.0f);
    k_scatter_add<<<ceil_div(E * nvec, NB), NB, 0, stream>>>(m, src, dst, a, E, C);
    k_cast_bf16<<<ceil_div(NNODES * C, NB), NB, 0, stream>>>(a, ab, NNODES * C);
    k_gemm_nt<<<gG, NB, 0, stream>>>(ab, wbIH, gi, NNODES, C, G3);
    k_gemm_nt<<<gG, NB, 0, stream>>>(hb, wbHH, gh, NNODES, C, G3);
    k_gru<<<ceil_div(NNODES * C, NB), NB, 0, stream>>>(gi, gh, bih, bhh, h, hn, NNODES, C);
    float* t = h; h = hn; hn = t;
  }
  *ph = h; *phn = hn;
}

// ---------------------------------------------------------------------------
extern "C" void kernel_launch(void* const* d_in, const int* in_sizes, int n_in,
                              void* d_out, int out_size, void* d_ws, size_t ws_size,
                              hipStream_t stream) {
  const float* x     = (const float*)d_in[0];
  const int*   eidx  = (const int*)d_in[1];
  const int*   batch = (const int*)d_in[2];
  const float* w1    = (const float*)d_in[3];
  const float* g1wih = (const float*)d_in[4];
  const float* g1whh = (const float*)d_in[5];
  const float* g1bih = (const float*)d_in[6];
  const float* g1bhh = (const float*)d_in[7];
  const float* w2    = (const float*)d_in[8];
  const float* g2wih = (const float*)d_in[9];
  const float* g2whh = (const float*)d_in[10];
  const float* g2bih = (const float*)d_in[11];
  const float* g2bhh = (const float*)d_in[12];
  const float* fcw   = (const float*)d_in[13];
  const float* fcb   = (const float*)d_in[14];
  float* out = (float*)d_out;

  const int E = in_sizes[1] / 2;
  const int* src = eidx;
  const int* dst = eidx + E;

  char* ws = (char*)d_ws;
  size_t off = 0;
  auto carve = [&](size_t bytes) -> void* {
    void* p = ws + off;
    off += (bytes + 255) & ~(size_t)255;
    return p;
  };

  float*  hA   = (float*)carve((size_t)NNODES * 256 * 4);
  float*  hB   = (float*)carve((size_t)NNODES * 256 * 4);
  float*  a    = (float*)carve((size_t)NNODES * 256 * 4);
  float*  m    = (float*)carve((size_t)NNODES * 256 * 4);
  float*  gi   = (float*)carve((size_t)NNODES * 768 * 4);
  float*  gh   = (float*)carve((size_t)NNODES * 768 * 4);
  __bf16* hb   = (__bf16*)carve((size_t)NNODES * 256 * 2);
  __bf16* ab   = (__bf16*)carve((size_t)NNODES * 256 * 2);
  __bf16* wbW  = (__bf16*)carve((size_t)3 * 256 * 256 * 2);
  __bf16* wbIH = (__bf16*)carve((size_t)256 * 768 * 2);
  __bf16* wbHH = (__bf16*)carve((size_t)256 * 768 * 2);
  float*  ymax = (float*)carve((size_t)NGRAPH * 256 * 4);
  (void)n_in; (void)out_size; (void)ws_size;

  const int NB = 256;

  // h0 = x  (phase 1 width 128)
  k_copy_f32<<<ceil_div(NNODES * 128, NB), NB, 0, stream>>>(x, hA, NNODES * 128);
  float* h = hA;
  float* hn = hB;

  // phase 1: d = 128
  run_phase(128, w1, g1wih, g1whh, g1bih, g1bhh, src, dst, E,
            &h, &hn, a, m, gi, gh, hb, ab, wbW, wbIH, wbHH, stream);

  // relu + zero-pad 128 -> 256
  k_relu_pad<<<ceil_div(NNODES * 256, NB), NB, 0, stream>>>(h, hn, NNODES);
  { float* t = h; h = hn; hn = t; }

  // phase 2: d = 256
  run_phase(256, w2, g2wih, g2whh, g2bih, g2bhh, src, dst, E,
            &h, &hn, a, m, gi, gh, hb, ab, wbW, wbIH, wbHH, stream);

  // global max pool per graph, then FC
  k_fill<<<ceil_div(NGRAPH * 256, NB), NB, 0, stream>>>(ymax, NGRAPH * 256,
                                                        -__builtin_inff());
  k_segmax<<<ceil_div(NNODES * 256, NB), NB, 0, stream>>>(h, batch, ymax, NNODES, 256);
  k_fc<<<ceil_div(NGRAPH * 6, NB), NB, 0, stream>>>(ymax, fcw, fcb, out);
}
